// MSDeformAttnTransformerEncoderOnly_2027224564177
// MI455X (gfx1250) — matrix operand order repaired and checked by reading, not compiled, not run on hardware
//
#include <hip/hip_runtime.h>
#include <hip/hip_bf16.h>
#include <cstddef>

// ---------------------------------------------------------------------------
// MS-Deformable-Attention transformer encoder + FPN fusion for MI455X
// (gfx1250). Dense math: V_WMMA_F32_16X16X32_F16, fp32 accumulate.
// Weights pre-converted once to f16 (N,K) layout and staged into LDS with
// GLOBAL_LOAD_ASYNC_TO_LDS_B128 (ASYNCcnt); activations converted fp32->f16
// in a double-buffered LDS pipeline so HBM latency overlaps the WMMA pipe.
// ---------------------------------------------------------------------------

typedef _Float16 v16h __attribute__((ext_vector_type(16)));
typedef _Float16 v8h  __attribute__((ext_vector_type(8)));
typedef float    v8f  __attribute__((ext_vector_type(8)));
typedef float    v4f  __attribute__((ext_vector_type(4)));
typedef int      v4i  __attribute__((ext_vector_type(4)));

// address-space-qualified vector pointers for the async builtin
using as1_v4i = __attribute__((address_space(1))) v4i;
using as3_v4i = __attribute__((address_space(3))) v4i;

#define BM 128
#define BN 128
#define BK 32

// ---- CDNA5 async global->LDS copy (16 bytes per lane) ----------------------
__device__ __forceinline__ void async_cp_b128(const _Float16* g, _Float16* l)
{
#if defined(__HIP_DEVICE_COMPILE__)
#if __has_builtin(__builtin_amdgcn_global_load_async_to_lds_b128)
    __builtin_amdgcn_global_load_async_to_lds_b128(
        (as1_v4i*)g, (as3_v4i*)l, 0, 0);
#else
    unsigned loff = (unsigned)(size_t)(__attribute__((address_space(3))) void*)l;
    asm volatile("global_load_async_to_lds_b128 %0, %1, off"
                 :: "v"(loff), "v"(g) : "memory");
#endif
#else
    (void)g; (void)l;
#endif
}

__device__ __forceinline__ void wait_async0()
{
#if defined(__HIP_DEVICE_COMPILE__)
#if __has_builtin(__builtin_amdgcn_s_wait_asynccnt)
    __builtin_amdgcn_s_wait_asynccnt(0);
#else
    asm volatile("s_wait_asynccnt 0" ::: "memory");
#endif
#endif
}

// ---- tile staging helpers --------------------------------------------------
// B tile: sB[n*32 + k] = Bt[(n0+n)*K + k0+k]; Bt is f16 (N,K) row-major.
// 512 16-byte lane-copies per tile -> 2 async instructions per wave.
__device__ __forceinline__ void stageB(const _Float16* __restrict__ Bt,
                                       _Float16* sBbuf, int n0, int k0,
                                       int N, int K, int tid)
{
    #pragma unroll
    for (int j = 0; j < 2; ++j) {
        int s  = tid + j * 256;        // 0..511
        int n  = s & 127;
        int kh = s >> 7;               // 0..3 (8 f16 each)
        int gn = n0 + n;
        _Float16* l = &sBbuf[n * BK + kh * 8];
        if (gn < N) {
            async_cp_b128(Bt + (size_t)gn * K + k0 + kh * 8, l);
        } else {
            v8h z = {};
            *(v8h*)l = z;              // keep WMMA clear of poisoned LDS
        }
    }
}

// A tile: issue fp32 global loads (optionally fused A+A2) ...
template<bool HAS_A2>
__device__ __forceinline__ void issueA(const float* __restrict__ A,
                                       const float* __restrict__ A2,
                                       int m0, int k0, int K, int tid, v4f r[4])
{
    #pragma unroll
    for (int it = 0; it < 4; ++it) {
        int p = tid + it * 256;        // float4 slot 0..1023
        int m = p >> 3;
        int k = (p & 7) << 2;
        size_t off = (size_t)(m0 + m) * K + k0 + k;
        v4f a = *(const v4f*)(A + off);
        if (HAS_A2) a = a + *(const v4f*)(A2 + off);
        r[it] = a;
    }
}
// ... and commit them (cvt fp32->f16 + ds_store) after the WMMAs have issued.
__device__ __forceinline__ void commitA(_Float16* sAbuf, int tid, const v4f r[4])
{
    #pragma unroll
    for (int it = 0; it < 4; ++it) {
        int p = tid + it * 256;
        int m = p >> 3;
        int k = (p & 7) << 2;
        _Float16* d = &sAbuf[m * BK + k];
        d[0] = (_Float16)r[it][0]; d[1] = (_Float16)r[it][1];
        d[2] = (_Float16)r[it][2]; d[3] = (_Float16)r[it][3];
    }
}

// 8 WMMAs on the current LDS tiles (contiguous k = 16*half + s per lane in both A and B).
__device__ __forceinline__ void compute_tile(const _Float16* sAbuf,
                                             const _Float16* sBbuf,
                                             int wm, int wn, int hl, int l16,
                                             v8f acc[2][4])
{
    v16h af[2];
    #pragma unroll
    for (int mt = 0; mt < 2; ++mt) {
        int m = wm * 32 + mt * 16 + l16;
        af[mt] = *(const v16h*)&sAbuf[m * BK + hl * 16];
    }
    v16h bf[4];
    #pragma unroll
    for (int nt = 0; nt < 4; ++nt) {
        int n = wn * 64 + nt * 16 + l16;
        bf[nt] = *(const v16h*)&sBbuf[n * BK + hl * 16];
    }
    #pragma unroll
    for (int mt = 0; mt < 2; ++mt)
        #pragma unroll
        for (int nt = 0; nt < 4; ++nt)
            acc[mt][nt] = __builtin_amdgcn_wmma_f32_16x16x32_f16(
                false, af[mt], false, bf[nt], (short)0, acc[mt][nt], false, false);
}

// ---------------------------------------------------------------------------
// GEMM: C[M,N] = (A [+A2])[M,K] * Bt^T + bias, optional ReLU.
// A fp32 row-major (lda=K), Bt f16 (N,K). M must be a multiple of 128.
// Double-buffered: stage tile t+1 while computing tile t.
// ---------------------------------------------------------------------------
template<bool HAS_A2, bool RELU>
__global__ __launch_bounds__(256)
void gemm_wmma_t(const float* __restrict__ A, const float* __restrict__ A2,
                 const _Float16* __restrict__ Bt, const float* __restrict__ bias,
                 float* __restrict__ C, int M, int N, int K)
{
    __shared__ __align__(32) _Float16 sA[2][BM * BK];
    __shared__ __align__(32) _Float16 sB[2][BN * BK];

    const int tid  = threadIdx.x;
    const int lane = tid & 31;
    const int wave = tid >> 5;
    const int hl   = lane >> 4;
    const int l16  = lane & 15;
    const int wm   = wave >> 1;
    const int wn   = wave & 1;
    const int m0   = blockIdx.y * BM;
    const int n0   = blockIdx.x * BN;

    v8f acc[2][4] = {};
    const int nk = K >> 5;

    v4f ar[4];
    issueA<HAS_A2>(A, A2, m0, 0, K, tid, ar);
    stageB(Bt, sB[0], n0, 0, N, K, tid);
    commitA(sA[0], tid, ar);
    wait_async0();
    __syncthreads();

    for (int t = 0; t < nk; ++t) {
        const int  cur  = t & 1;
        const bool more = (t + 1 < nk);
        if (more) {
            issueA<HAS_A2>(A, A2, m0, (t + 1) << 5, K, tid, ar);
            stageB(Bt, sB[cur ^ 1], n0, (t + 1) << 5, N, K, tid);
        }
        compute_tile(sA[cur], sB[cur], wm, wn, hl, l16, acc);
        if (more) commitA(sA[cur ^ 1], tid, ar);
        wait_async0();
        __syncthreads();
    }

    #pragma unroll
    for (int mt = 0; mt < 2; ++mt) {
        #pragma unroll
        for (int nt = 0; nt < 4; ++nt) {
            int gn = n0 + wn * 64 + nt * 16 + l16;
            if (gn >= N) continue;
            float bb = bias ? bias[gn] : 0.f;
            #pragma unroll
            for (int r = 0; r < 8; ++r) {
                int gm = m0 + wm * 32 + mt * 16 + hl * 8 + r;
                float v = acc[mt][nt][r] + bb;
                if (RELU) v = fmaxf(v, 0.f);
                C[(size_t)gm * N + gn] = v;
            }
        }
    }
}

// ---------------------------------------------------------------------------
// 3x3 conv as implicit-GEMM WMMA. Y: (BT,128,128,256) NHWC fp32.
// Wc: f16 (Cout, 9*256) with k = tap*256 + cin. K = 2304.
// mask_features (B,C,T,128,128) += relu(conv + bias). Same pipeline.
// ---------------------------------------------------------------------------
__device__ __forceinline__ void issueA_conv(const float* __restrict__ Y,
                                            int m0, int k0, int tid, v4f r[4])
{
    const int o  = k0 >> 8;            // tap 0..8 (32 | 256)
    const int dy = o / 3 - 1;
    const int dx = o % 3 - 1;
    const int c0 = k0 & 255;
    #pragma unroll
    for (int it = 0; it < 4; ++it) {
        int p  = tid + it * 256;
        int m  = p >> 3;
        int k  = (p & 7) << 2;
        int gm = m0 + m;
        int bt = gm >> 14;
        int pp = gm & 16383;
        int py = (pp >> 7) + dy;
        int px = (pp & 127) + dx;
        v4f a = {};
        if (py >= 0 && py < 128 && px >= 0 && px < 128)
            a = *(const v4f*)(Y + (((size_t)(bt * 128 + py) * 128 + px) * 256
                                   + c0 + k));
        r[it] = a;
    }
}

__global__ __launch_bounds__(256)
void conv3x3_wmma(const float* __restrict__ Y, const _Float16* __restrict__ Wc,
                  const float* __restrict__ bias, float* __restrict__ MF)
{
    __shared__ __align__(32) _Float16 sA[2][BM * BK];
    __shared__ __align__(32) _Float16 sB[2][BN * BK];

    const int tid  = threadIdx.x;
    const int lane = tid & 31;
    const int wave = tid >> 5;
    const int hl   = lane >> 4;
    const int l16  = lane & 15;
    const int wm   = wave >> 1;
    const int wn   = wave & 1;
    const int m0   = blockIdx.y * BM;
    const int n0   = blockIdx.x * BN;

    v8f acc[2][4] = {};
    const int nk = 2304 / 32;          // 72

    v4f ar[4];
    issueA_conv(Y, m0, 0, tid, ar);
    stageB(Wc, sB[0], n0, 0, 256, 2304, tid);
    commitA(sA[0], tid, ar);
    wait_async0();
    __syncthreads();

    for (int t = 0; t < nk; ++t) {
        const int  cur  = t & 1;
        const bool more = (t + 1 < nk);
        if (more) {
            issueA_conv(Y, m0, (t + 1) << 5, tid, ar);
            stageB(Wc, sB[cur ^ 1], n0, (t + 1) << 5, 256, 2304, tid);
        }
        compute_tile(sA[cur], sB[cur], wm, wn, hl, l16, acc);
        if (more) commitA(sA[cur ^ 1], tid, ar);
        wait_async0();
        __syncthreads();
    }

    #pragma unroll
    for (int mt = 0; mt < 2; ++mt) {
        #pragma unroll
        for (int nt = 0; nt < 4; ++nt) {
            int gn = n0 + wn * 64 + nt * 16 + l16;
            float bb = bias[gn];
            #pragma unroll
            for (int r = 0; r < 8; ++r) {
                int gm = m0 + wm * 32 + mt * 16 + hl * 8 + r;
                int bt = gm >> 14;
                int b  = bt >> 1;
                int t_ = bt & 1;
                int pp = gm & 16383;
                float v = fmaxf(acc[mt][nt][r] + bb, 0.f);
                size_t idx = (((size_t)b * 256 + gn) * 2 + t_) * 16384 + pp;
                MF[idx] += v;
            }
        }
    }
}

// ---------------------------------------------------------------------------
// Weight pre-conversion kernels (run once per call, ~16.6 MB f16 total).
// ---------------------------------------------------------------------------
// (L,K,N) fp32 -> (L,N,K) f16
__global__ __launch_bounds__(256)
void wcvt_kn(const float* __restrict__ src, _Float16* __restrict__ dst,
             int L, int N, int K)
{
    size_t i = (size_t)blockIdx.x * 256 + threadIdx.x;
    size_t tot = (size_t)L * N * K;
    if (i >= tot) return;
    int nk = N * K;
    int l = (int)(i / nk), r = (int)(i % nk);
    int n = r / K, k = r % K;
    dst[i] = (_Float16)src[(size_t)l * nk + (size_t)k * N + n];
}
// straight copy-convert (lat_w already (L,N,K))
__global__ __launch_bounds__(256)
void wcvt_cp(const float* __restrict__ src, _Float16* __restrict__ dst, size_t tot)
{
    size_t i = (size_t)blockIdx.x * 256 + threadIdx.x;
    if (i < tot) dst[i] = (_Float16)src[i];
}
// outc_w (L,Cout,Cin,3,3) -> (L,Cout, 9*256) f16 with k = tap*256 + cin
__global__ __launch_bounds__(256)
void wcvt_outc(const float* __restrict__ src, _Float16* __restrict__ dst)
{
    size_t i = (size_t)blockIdx.x * 256 + threadIdx.x;
    size_t tot = (size_t)6 * 256 * 2304;
    if (i >= tot) return;
    int l  = (int)(i / (256 * 2304));
    int r  = (int)(i % (256 * 2304));
    int n  = r / 2304;
    int rr = r % 2304;
    int o  = rr / 256;
    int c  = rr % 256;
    dst[i] = (_Float16)src[((((size_t)l * 256 + n) * 256 + c) * 9) + o];
}

// ---------------------------------------------------------------------------
// Deformable-attention sampling. One wave per (bt, q, head); lanes = head dim.
// ---------------------------------------------------------------------------
__global__ __launch_bounds__(256)
void msdeform_sample(const float* __restrict__ V, const float* __restrict__ OFF,
                     const float* __restrict__ AW, float* __restrict__ ACC)
{
    const int gw = blockIdx.x * 8 + (threadIdx.x >> 5);
    const int d  = threadIdx.x & 31;
    const int h  = gw & 7;
    const int q  = (gw >> 3) % 5376;
    const int bt = gw / (5376 * 8);

    const float* awrow = AW + ((size_t)bt * 5376 + q) * 96 + h * 12;
    float w[12];
    float mx = -1e30f;
    #pragma unroll
    for (int j = 0; j < 12; ++j) { w[j] = awrow[j]; mx = fmaxf(mx, w[j]); }
    float s = 0.f;
    #pragma unroll
    for (int j = 0; j < 12; ++j) { w[j] = __expf(w[j] - mx); s += w[j]; }
    const float inv = 1.f / s;

    int qq, Wl;
    if (q < 256)        { qq = q;        Wl = 16; }
    else if (q < 1280)  { qq = q - 256;  Wl = 32; }
    else                { qq = q - 1280; Wl = 64; }
    const float rx = ((qq % Wl) + 0.5f) / (float)Wl;
    const float ry = ((qq / Wl) + 0.5f) / (float)Wl;

    const float* offrow = OFF + ((size_t)bt * 5376 + q) * 192 + h * 24;
    const int starts[3] = {0, 256, 1280};
    const int sz[3]     = {16, 32, 64};

    float accv = 0.f;
    #pragma unroll
    for (int l = 0; l < 3; ++l) {
        const int HW = sz[l];
        const float fHW = (float)HW;
        const float* vb = V + ((size_t)bt * 5376 + starts[l]) * 256 + h * 32 + d;
        #pragma unroll
        for (int p = 0; p < 4; ++p) {
            float ox = offrow[(l * 4 + p) * 2 + 0];
            float oy = offrow[(l * 4 + p) * 2 + 1];
            float x = (rx + ox / fHW) * fHW - 0.5f;
            float y = (ry + oy / fHW) * fHW - 0.5f;
            float xf = floorf(x), yf = floorf(y);
            int x0 = (int)xf, y0 = (int)yf;
            float fx = x - xf, fy = y - yf;
            bool vx0 = (x0 >= 0)     && (x0 < HW);
            bool vx1 = (x0 + 1 >= 0) && (x0 + 1 < HW);
            bool vy0 = (y0 >= 0)     && (y0 < HW);
            bool vy1 = (y0 + 1 >= 0) && (y0 + 1 < HW);
            float v00 = (vx0 && vy0) ? vb[(size_t)(y0 * HW + x0) * 256]           : 0.f;
            float v01 = (vx1 && vy0) ? vb[(size_t)(y0 * HW + x0 + 1) * 256]       : 0.f;
            float v10 = (vx0 && vy1) ? vb[(size_t)((y0 + 1) * HW + x0) * 256]     : 0.f;
            float v11 = (vx1 && vy1) ? vb[(size_t)((y0 + 1) * HW + x0 + 1) * 256] : 0.f;
            float bil = v00 * (1.f - fx) * (1.f - fy) + v01 * fx * (1.f - fy)
                      + v10 * (1.f - fx) * fy         + v11 * fx * fy;
            accv += w[l * 4 + p] * inv * bil;
        }
    }
    ACC[((size_t)bt * 5376 + q) * 256 + h * 32 + d] = accv;
}

// ---------------------------------------------------------------------------
// Residual + LayerNorm over C=256 (one block per row): X = LN(X + D)*g + b
// ---------------------------------------------------------------------------
__global__ __launch_bounds__(256)
void ln_residual(float* __restrict__ X, const float* __restrict__ D,
                 const float* __restrict__ g, const float* __restrict__ b)
{
    __shared__ float red[256];
    const int row = blockIdx.x;
    const int t   = threadIdx.x;
    const size_t base = (size_t)row * 256;
    float v = X[base + t] + D[base + t];
    red[t] = v;
    __syncthreads();
    for (int s = 128; s > 0; s >>= 1) { if (t < s) red[t] += red[t + s]; __syncthreads(); }
    const float mean = red[0] * (1.f / 256.f);
    __syncthreads();
    const float dv = v - mean;
    red[t] = dv * dv;
    __syncthreads();
    for (int s = 128; s > 0; s >>= 1) { if (t < s) red[t] += red[t + s]; __syncthreads(); }
    const float var = red[0] * (1.f / 256.f);
    X[base + t] = dv * rsqrtf(var + 1e-5f) * g[t] + b[t];
}

// ---------------------------------------------------------------------------
// Build output / posf states from per-level (BT,C,H,W) sources.
// ---------------------------------------------------------------------------
__global__ __launch_bounds__(256)
void build_io(const float* __restrict__ s0, const float* __restrict__ p0,
              const float* __restrict__ s1, const float* __restrict__ p1,
              const float* __restrict__ s2, const float* __restrict__ p2,
              const float* __restrict__ lev,
              float* __restrict__ OUT, float* __restrict__ POSF)
{
    const int idx = blockIdx.x;          // bt*5376 + q
    const int c   = threadIdx.x;
    const int bt  = idx / 5376;
    const int q   = idx % 5376;
    const float *s, *pp; int l, hw, qq;
    if (q < 256)        { l = 0; hw = 256;  qq = q;        s = s0; pp = p0; }
    else if (q < 1280)  { l = 1; hw = 1024; qq = q - 256;  s = s1; pp = p1; }
    else                { l = 2; hw = 4096; qq = q - 1280; s = s2; pp = p2; }
    const size_t si = ((size_t)bt * 256 + c) * hw + qq;
    OUT[(size_t)idx * 256 + c]  = s[si];
    POSF[(size_t)idx * 256 + c] = pp[si] + lev[l * 256 + c];
}

// (B,C,T,HW) -> (BT*HW, C) NHWC for the lateral 1x1-conv GEMM
__global__ __launch_bounds__(256)
void mf_to_nhwc(const float* __restrict__ MF, float* __restrict__ XT)
{
    const int m  = blockIdx.x;
    const int c  = threadIdx.x;
    const int bt = m >> 14;
    const int b  = bt >> 1;
    const int t  = bt & 1;
    const int pp = m & 16383;
    XT[(size_t)m * 256 + c] = MF[(((size_t)b * 256 + c) * 2 + t) * 16384 + pp];
}

// Y (BT,128,128,C NHWC) += bilinear-upsample(finest 64x64 slice of OUT)
__global__ __launch_bounds__(256)
void upsample_add(const float* __restrict__ OUT, float* __restrict__ Y)
{
    const int m  = blockIdx.x;
    const int c  = threadIdx.x;
    const int bt = m >> 14;
    const int pp = m & 16383;
    const int py = pp >> 7, px = pp & 127;
    const float iy = py * 0.5f - 0.25f;
    const float ix = px * 0.5f - 0.25f;
    const int y0 = (int)floorf(iy), x0 = (int)floorf(ix);
    const float fy = iy - (float)y0, fx = ix - (float)x0;
    int y0c = y0 < 0 ? 0 : (y0 > 63 ? 63 : y0);
    int y1c = (y0 + 1) < 0 ? 0 : ((y0 + 1) > 63 ? 63 : y0 + 1);
    int x0c = x0 < 0 ? 0 : (x0 > 63 ? 63 : x0);
    int x1c = (x0 + 1) < 0 ? 0 : ((x0 + 1) > 63 ? 63 : x0 + 1);
    const float* base = OUT + ((size_t)bt * 5376 + 1280) * 256 + c;
    float v00 = base[(size_t)(y0c * 64 + x0c) * 256];
    float v01 = base[(size_t)(y0c * 64 + x1c) * 256];
    float v10 = base[(size_t)(y1c * 64 + x0c) * 256];
    float v11 = base[(size_t)(y1c * 64 + x1c) * 256];
    float v = v00 * (1.f - fy) * (1.f - fx) + v01 * (1.f - fy) * fx
            + v10 * fy * (1.f - fx)         + v11 * fy * fx;
    Y[(size_t)m * 256 + c] += v;
}

// ---------------------------------------------------------------------------
// Host-side orchestration
// ---------------------------------------------------------------------------
extern "C" void kernel_launch(void* const* d_in, const int* in_sizes, int n_in,
                              void* d_out, int out_size, void* d_ws, size_t ws_size,
                              hipStream_t stream)
{
    (void)in_sizes; (void)n_in; (void)out_size; (void)ws_size;

    const float* src0 = (const float*)d_in[0];
    const float* pos0 = (const float*)d_in[1];
    const float* src1 = (const float*)d_in[2];
    const float* pos1 = (const float*)d_in[3];
    const float* src2 = (const float*)d_in[4];
    const float* pos2 = (const float*)d_in[5];
    const float* mask_features = (const float*)d_in[6];
    const float* level_embed   = (const float*)d_in[7];
    const float* off_w  = (const float*)d_in[8];
    const float* off_b  = (const float*)d_in[9];
    const float* aw_w   = (const float*)d_in[10];
    const float* aw_b   = (const float*)d_in[11];
    const float* val_w  = (const float*)d_in[12];
    const float* val_b  = (const float*)d_in[13];
    const float* out_w  = (const float*)d_in[14];
    const float* out_b  = (const float*)d_in[15];
    const float* ln1_g  = (const float*)d_in[16];
    const float* ln1_b  = (const float*)d_in[17];
    const float* ff1_w  = (const float*)d_in[18];
    const float* ff1_b  = (const float*)d_in[19];
    const float* ff2_w  = (const float*)d_in[20];
    const float* ff2_b  = (const float*)d_in[21];
    const float* ln2_g  = (const float*)d_in[22];
    const float* ln2_b  = (const float*)d_in[23];
    const float* lat_w  = (const float*)d_in[24];
    const float* lat_b  = (const float*)d_in[25];
    const float* outc_w = (const float*)d_in[26];
    const float* outc_b = (const float*)d_in[27];

    constexpr int    BT = 4, C = 256, LQ = 5376, DFF = 1024;
    constexpr size_t OUT_N = (size_t)BT * LQ * C;
    constexpr size_t MF_N  = (size_t)2 * C * 2 * 16384;
    constexpr size_t OFF_N = (size_t)BT * LQ * 192;
    constexpr size_t AW_N  = (size_t)BT * LQ * 96;
    constexpr size_t FFH_N = (size_t)BT * LQ * DFF;
    constexpr size_t XT_N  = (size_t)BT * 16384 * C;

    float* ws = (float*)d_ws;
    size_t o = 0;
    float* OUT  = ws + o; o += OUT_N;
    float* POSF = ws + o; o += OUT_N;
    float* Vv   = ws + o; o += OUT_N;
    float* ACCb = ws + o; o += OUT_N;
    float* ATTN = ws + o; o += OUT_N;
    float* MF   = ws + o; o += MF_N;
    float* XT   = ws + o; o += XT_N;
    float* LATY = ws + o; o += XT_N;
    float* OFFb = ws + o; o += OFF_N;
    float* AWb  = ws + o; o += AW_N;
    float* FFH  = ws + o; o += FFH_N;

    // f16 weight scratch (16B-aligned; all sub-sizes are multiples of 8)
    o = (o + 3) & ~(size_t)3;
    _Float16* hw = (_Float16*)(ws + o);
    size_t ho = 0;
    _Float16* hval  = hw + ho; ho += (size_t)6 * 65536;
    _Float16* hout  = hw + ho; ho += (size_t)6 * 65536;
    _Float16* hoff  = hw + ho; ho += (size_t)6 * 49152;
    _Float16* haw   = hw + ho; ho += (size_t)6 * 24576;
    _Float16* hff1  = hw + ho; ho += (size_t)6 * 262144;
    _Float16* hff2  = hw + ho; ho += (size_t)6 * 262144;
    _Float16* hlat  = hw + ho; ho += (size_t)6 * 65536;
    _Float16* houtc = hw + ho; ho += (size_t)6 * 589824;

    const dim3 blk(256);
    auto ggrid = [](int M, int N) { return dim3((N + 127) / 128, (M + 127) / 128); };
    auto cgrid = [](size_t n) { return dim3((unsigned)((n + 255) / 256)); };
    const int M = BT * LQ;              // 21504

    // ---- one-time per call: weights -> f16 (N,K) ----
    wcvt_kn<<<cgrid((size_t)6 * 65536),  blk, 0, stream>>>(val_w, hval, 6, 256, 256);
    wcvt_kn<<<cgrid((size_t)6 * 65536),  blk, 0, stream>>>(out_w, hout, 6, 256, 256);
    wcvt_kn<<<cgrid((size_t)6 * 49152),  blk, 0, stream>>>(off_w, hoff, 6, 192, 256);
    wcvt_kn<<<cgrid((size_t)6 * 24576),  blk, 0, stream>>>(aw_w,  haw,  6, 96,  256);
    wcvt_kn<<<cgrid((size_t)6 * 262144), blk, 0, stream>>>(ff1_w, hff1, 6, 1024, 256);
    wcvt_kn<<<cgrid((size_t)6 * 262144), blk, 0, stream>>>(ff2_w, hff2, 6, 256, 1024);
    wcvt_cp<<<cgrid((size_t)6 * 65536),  blk, 0, stream>>>(lat_w, hlat, (size_t)6 * 65536);
    wcvt_outc<<<cgrid((size_t)6 * 589824), blk, 0, stream>>>(outc_w, houtc);

    build_io<<<dim3(BT * LQ), blk, 0, stream>>>(src0, pos0, src1, pos1, src2, pos2,
                                                level_embed, OUT, POSF);
    (void)hipMemcpyAsync(MF, mask_features, MF_N * sizeof(float),
                         hipMemcpyDeviceToDevice, stream);

    for (int i = 0; i < 6; ++i) {
        gemm_wmma_t<false, false><<<ggrid(M, 256), blk, 0, stream>>>(
            OUT, nullptr, hval + (size_t)i * 65536, val_b + i * 256, Vv, M, 256, 256);
        gemm_wmma_t<true, false><<<ggrid(M, 192), blk, 0, stream>>>(
            OUT, POSF, hoff + (size_t)i * 49152, off_b + i * 192, OFFb, M, 192, 256);
        gemm_wmma_t<true, false><<<ggrid(M, 96), blk, 0, stream>>>(
            OUT, POSF, haw + (size_t)i * 24576, aw_b + i * 96, AWb, M, 96, 256);
        msdeform_sample<<<dim3(BT * LQ), blk, 0, stream>>>(Vv, OFFb, AWb, ACCb);
        gemm_wmma_t<false, false><<<ggrid(M, 256), blk, 0, stream>>>(
            ACCb, nullptr, hout + (size_t)i * 65536, out_b + i * 256, ATTN, M, 256, 256);
        ln_residual<<<dim3(M), blk, 0, stream>>>(OUT, ATTN, ln1_g + i * 256, ln1_b + i * 256);
        gemm_wmma_t<false, true><<<ggrid(M, 1024), blk, 0, stream>>>(
            OUT, nullptr, hff1 + (size_t)i * 262144, ff1_b + i * 1024, FFH, M, 1024, 256);
        gemm_wmma_t<false, false><<<ggrid(M, 256), blk, 0, stream>>>(
            FFH, nullptr, hff2 + (size_t)i * 262144, ff2_b + i * 256, ATTN, M, 256, 1024);
        ln_residual<<<dim3(M), blk, 0, stream>>>(OUT, ATTN, ln2_g + i * 256, ln2_b + i * 256);

        mf_to_nhwc<<<dim3(BT * 16384), blk, 0, stream>>>(MF, XT);
        gemm_wmma_t<false, false><<<ggrid(BT * 16384, 256), blk, 0, stream>>>(
            XT, nullptr, hlat + (size_t)i * 65536, lat_b + i * 256, LATY, BT * 16384, 256, 256);
        upsample_add<<<dim3(BT * 16384), blk, 0, stream>>>(OUT, LATY);
        conv3x3_wmma<<<dim3(2, 512), blk, 0, stream>>>(
            LATY, houtc + (size_t)i * 589824, outc_b + i * 256, MF);
    }

    (void)hipMemcpyAsync(d_out, OUT, OUT_N * sizeof(float),
                         hipMemcpyDeviceToDevice, stream);
    (void)hipMemcpyAsync((float*)d_out + OUT_N, MF, MF_N * sizeof(float),
                         hipMemcpyDeviceToDevice, stream);
}
